// MultiHeadSelfAttention_15805479649382
// MI455X (gfx1250) — compile-verified, matrix operand order checked by reading, not run on hardware
//
#include <hip/hip_runtime.h>

typedef __bf16 bf16;
typedef __attribute__((ext_vector_type(4)))  __bf16 v4bf;
typedef __attribute__((ext_vector_type(8)))  __bf16 v8bf;
typedef __attribute__((ext_vector_type(16))) __bf16 v16bf;
typedef __attribute__((ext_vector_type(8)))  float  v8f;
typedef __attribute__((ext_vector_type(4)))  unsigned int u32x4;
typedef __attribute__((ext_vector_type(4)))  int i32x4;
typedef __attribute__((ext_vector_type(8)))  int i32x8;

#define B_ 2
#define S_ 2048
#define D_ 1024
#define H_ 16
#define HD_ 64

// ---------------------------------------------------------------------------
// CDNA5 feature detection (device pass only; host pass takes fallbacks)
// ---------------------------------------------------------------------------
#if defined(__HIP_DEVICE_COMPILE__) && __has_builtin(__builtin_amdgcn_global_load_async_to_lds_b128)
#define HAS_ASYNC_LDS 1
#else
#define HAS_ASYNC_LDS 0
#endif

#if defined(__HIP_DEVICE_COMPILE__) && __has_builtin(__builtin_amdgcn_tensor_load_to_lds) && __has_builtin(__builtin_amdgcn_s_wait_tensorcnt)
#define HAS_TDM 1
#else
#define HAS_TDM 0
#endif

#if HAS_ASYNC_LDS
typedef __attribute__((address_space(1))) i32x4 gas_i32x4;  // global
typedef __attribute__((address_space(3))) i32x4 las_i32x4;  // LDS
#endif

// copy 16B global -> LDS (async on CDNA5, tracked by ASYNCcnt)
__device__ __forceinline__ void cp16(bf16* l, const bf16* g) {
#if HAS_ASYNC_LDS
    __builtin_amdgcn_global_load_async_to_lds_b128(
        (gas_i32x4*)(unsigned long long)g,
        (las_i32x4*)(unsigned)(unsigned long long)l,
        0, 0);
#else
    *(v8bf*)l = *(const v8bf*)g;
#endif
}

template <int N>
__device__ __forceinline__ void async_wait() {
#if HAS_ASYNC_LDS
#if __has_builtin(__builtin_amdgcn_s_wait_asynccnt)
    __builtin_amdgcn_s_wait_asynccnt(N);
#else
    asm volatile("s_wait_asynccnt %0" :: "n"(N) : "memory");
#endif
#endif
}

#if HAS_TDM
// TDM: DMA one 64x64 bf16 tile of a [2048 x 64] row-major tensor into LDS.
// D# group0: count=1, lds_addr, 57-bit global_addr, type=2 ("image").
// D# group1: data_size=2B, tensor_dim0=64, tensor_dim1=2048, tile 64x64,
//            tensor_dim0_stride=64.
__device__ __forceinline__ void tdm_load_tile64x64(unsigned ldsOff,
                                                   unsigned long long ga) {
    u32x4 g0 = {};
    g0[0] = 1u;                                        // count = 1 valid D#
    g0[1] = ldsOff;                                    // lds_addr (bytes)
    g0[2] = (unsigned)ga;                              // global_addr[31:0]
    g0[3] = (unsigned)((ga >> 32) & 0x1FFFFFFull) | (2u << 30);  // [56:32] | type=2
    i32x8 g1 = {};
    g1[0] = (int)(1u << 16);                           // data_size = 1 -> 2 bytes
    g1[1] = (int)(64u << 16);                          // tensor_dim0 lo16
    g1[2] = (int)(2048u << 16);                        // dim0 hi16=0 | tensor_dim1 lo16
    g1[3] = (int)(64u << 16);                          // dim1 hi16=0 | tile_dim0=64
    g1[4] = 64;                                        // tile_dim1=64 | tile_dim2=0
    g1[5] = 64;                                        // tensor_dim0_stride lo32
    i32x4 z4 = {};
    i32x8 z8 = {};
#if __clang_major__ >= 23
    __builtin_amdgcn_tensor_load_to_lds(g0, g1, z4, z4, z8, 0);
    (void)z8;
#else
    __builtin_amdgcn_tensor_load_to_lds(g0, g1, z4, z4, 0);
    (void)z8;
#endif
}
#endif

// ---------------------------------------------------------------------------
// operand fragment loader: 16x32 bf16 A/B slice, row-major source.
// lanes 0-15 (row=lane) hold K=[0..7],[16..23]; lanes 16-31 hold K=[8..15],[24..31].
// caller passes p = &tile[row*stride + kchunk + (lane>>4)*8].
// ---------------------------------------------------------------------------
__device__ __forceinline__ v16bf ld_frag(const bf16* p) {
    v8bf lo = *(const v8bf*)(p);
    v8bf hi = *(const v8bf*)(p + 16);
    v16bf r;
#pragma unroll
    for (int i = 0; i < 8; ++i) { r[i] = lo[i]; r[8 + i] = hi[i]; }
    return r;
}

__device__ __forceinline__ v8f wmma_bf16(v16bf a, v16bf b, v8f c) {
    return __builtin_amdgcn_wmma_f32_16x16x32_bf16(false, a, false, b,
                                                   (short)0, c, false, false);
}

// ---------------------------------------------------------------------------
// fp32 -> bf16 conversion
// ---------------------------------------------------------------------------
__global__ void __launch_bounds__(256) cvt_f32_bf16(const float* __restrict__ src,
                                                    bf16* __restrict__ dst, int n4) {
    int i = blockIdx.x * blockDim.x + threadIdx.x;
    if (i >= n4) return;
    float4 f = ((const float4*)src)[i];
    v4bf o;
    o[0] = (bf16)f.x; o[1] = (bf16)f.y; o[2] = (bf16)f.z; o[3] = (bf16)f.w;
    ((v4bf*)dst)[i] = o;
}

// ---------------------------------------------------------------------------
// GEMM: C[m][n] = sum_k A[m*K+k] * W[n*K+k];  A: MxK, W: NxK, both row-major.
// 64x64 tile/block, 4 waves 2x2, K-step 64, double-buffered async LDS fills.
// ---------------------------------------------------------------------------
template <bool SPLIT_HEADS, bool ADD_BIAS>
__global__ void __launch_bounds__(128)
gemm_bf16(const bf16* __restrict__ A, const bf16* __restrict__ W,
          const float* __restrict__ bias, void* __restrict__ out,
          int M, int N, int K) {
    __shared__ bf16 sA[2][64 * 64];
    __shared__ bf16 sW[2][64 * 64];
    const int t = threadIdx.x;
    const int lane = t & 31, wv = t >> 5;
    const int lrow = lane & 15, hi = lane >> 4, lk = hi * 8;
    const int m0 = blockIdx.x * 64, n0 = blockIdx.y * 64;
    const int wm = (wv >> 1) * 32, wn = (wv & 1) * 32;
    v8f acc[2][2] = {};

    const int sr = t >> 1;           // staging row 0..63
    const int sc = (t & 1) * 32;     // staging col base (64B per thread per tile)

    auto stage = [&](int buf, int k0) {
        const bf16* ga = A + (size_t)(m0 + sr) * K + k0 + sc;
        const bf16* gw = W + (size_t)(n0 + sr) * K + k0 + sc;
#pragma unroll
        for (int u = 0; u < 4; ++u) {
            cp16(&sA[buf][sr * 64 + sc + u * 8], ga + u * 8);
            cp16(&sW[buf][sr * 64 + sc + u * 8], gw + u * 8);
        }
    };

    stage(0, 0);
    int buf = 0;
    for (int k0 = 0; k0 < K; k0 += 64) {
        if (k0 + 64 < K) {            // prefetch next stage, overlap with wait
            stage(buf ^ 1, k0 + 64);
            async_wait<8>();          // previous 8 fills done; 8 new in flight
        } else {
            async_wait<0>();
        }
        __syncthreads();

        const bf16* cA = sA[buf];
        const bf16* cW = sW[buf];
#pragma unroll
        for (int kc = 0; kc < 2; ++kc) {
            v16bf af[2], wf[2];
#pragma unroll
            for (int i = 0; i < 2; ++i)
                af[i] = ld_frag(cA + (wm + i * 16 + lrow) * 64 + kc * 32 + lk);
#pragma unroll
            for (int j = 0; j < 2; ++j)
                wf[j] = ld_frag(cW + (wn + j * 16 + lrow) * 64 + kc * 32 + lk);
#pragma unroll
            for (int i = 0; i < 2; ++i)
#pragma unroll
                for (int j = 0; j < 2; ++j)
                    acc[i][j] = wmma_bf16(af[i], wf[j], acc[i][j]);
        }
        __syncthreads();
        buf ^= 1;
    }

    // epilogue: C layout -> lane holds (rows r+8*hi, col lane%16) per 16x16 tile
#pragma unroll
    for (int i = 0; i < 2; ++i)
#pragma unroll
        for (int j = 0; j < 2; ++j)
#pragma unroll
            for (int r = 0; r < 8; ++r) {
                int mg = m0 + wm + i * 16 + r + 8 * hi;
                int ng = n0 + wn + j * 16 + lrow;
                float c = acc[i][j][r];
                if (SPLIT_HEADS) {
                    int b = mg >> 11, sidx = mg & (S_ - 1);
                    int h = ng >> 6,  hd   = ng & (HD_ - 1);
                    ((bf16*)out)[(((size_t)(b * H_ + h) * S_) + sidx) * HD_ + hd] = (bf16)c;
                } else {
                    float bv = ADD_BIAS ? bias[ng] : 0.f;
                    ((float*)out)[(size_t)mg * N + ng] = c + bv;
                }
            }
}

// ---------------------------------------------------------------------------
// Flash attention (causal).  Q/K/V bf16 [B,H,S,HD].  grid=(S/64, B*H), 128 thr.
// K tile staged via TDM (tensor_load_to_lds) when available; V staged
// transposed; per-wave online softmax; WMMA for Q·K^T and P·V.
// ---------------------------------------------------------------------------
__global__ void __launch_bounds__(128)
flash_attn(const bf16* __restrict__ Q, const bf16* __restrict__ Km,
           const bf16* __restrict__ V, bf16* __restrict__ ctx) {
    __shared__ bf16 sK [64 * 64];      // [key][hd]
    __shared__ bf16 sVt[64 * 64];      // [hd][key]
    __shared__ bf16 sP [4 * 16 * 64];  // per-wave P tiles [row][key]

    const int t = threadIdx.x, lane = t & 31, wv = t >> 5;
    const int lrow = lane & 15, hi = lane >> 4, lk = hi * 8;
    const int bh = blockIdx.y;
    const int qb = blockIdx.x * 64;
    const size_t base = (size_t)bh * S_ * HD_;

    v16bf qf[2];
    {
        const bf16* qp = Q + base + (size_t)(qb + wv * 16 + lrow) * HD_;
        qf[0] = ld_frag(qp + lk);
        qf[1] = ld_frag(qp + 32 + lk);
    }

    v8f acc[4] = {};
    float mrun[8], lrun[8];
#pragma unroll
    for (int r = 0; r < 8; ++r) { mrun[r] = -1e30f; lrun[r] = 0.f; }
    bf16* myP = sP + wv * (16 * 64);

    const int sr = t >> 1, sc0 = (t & 1) * 32;

    for (int kv = 0; kv <= qb; kv += 64) {
        // ---- stage K tile: TDM DMA (one wave issues) or async/sync fills ----
#if HAS_TDM
        if (t < 32)
            tdm_load_tile64x64((unsigned)(unsigned long long)sK,
                               (unsigned long long)(Km + base + (size_t)kv * HD_));
#else
        {
            const bf16* kp = Km + base + (size_t)(kv + sr) * HD_ + sc0;
#pragma unroll
            for (int u = 0; u < 4; ++u)
                cp16(sK + sr * 64 + sc0 + u * 8, kp + u * 8);
        }
#endif
        // ---- stage V transposed (overlaps the K DMA) ----
        {
            const bf16* vp = V + base + (size_t)(kv + sr) * HD_ + sc0;
            v8bf vv[4];
#pragma unroll
            for (int u = 0; u < 4; ++u) vv[u] = *(const v8bf*)(vp + u * 8);
#pragma unroll
            for (int u = 0; u < 32; ++u)
                sVt[(sc0 + u) * 64 + sr] = vv[u >> 3][u & 7];
        }
#if HAS_TDM
        if (t < 32) __builtin_amdgcn_s_wait_tensorcnt(0);
#else
        async_wait<0>();
#endif
        __syncthreads();

        // ---- S = Q·K^T ----
        v8f s[4];
#pragma unroll
        for (int j = 0; j < 4; ++j) {
            v8f a = {};
            v16bf b0 = ld_frag(sK + (j * 16 + lrow) * 64 + lk);
            v16bf b1 = ld_frag(sK + (j * 16 + lrow) * 64 + 32 + lk);
            a = wmma_bf16(qf[0], b0, a);
            a = wmma_bf16(qf[1], b1, a);
            s[j] = a;
        }

        // ---- causal mask + scale + online softmax ----
#pragma unroll
        for (int r = 0; r < 8; ++r) {
            const int rg = qb + wv * 16 + r + 8 * hi;
            float rm = -1e30f;
#pragma unroll
            for (int j = 0; j < 4; ++j) {
                int cg = kv + j * 16 + lrow;
                float x = (cg > rg) ? -1e30f : s[j][r] * 0.125f;  // 1/sqrt(64)
                s[j][r] = x;
                rm = fmaxf(rm, x);
            }
#pragma unroll
            for (int off = 1; off < 16; off <<= 1)
                rm = fmaxf(rm, __shfl_xor(rm, off, 32));
            float mnew = fmaxf(mrun[r], rm);
            float alpha = __expf(mrun[r] - mnew);
            mrun[r] = mnew;
            float rsum = 0.f;
#pragma unroll
            for (int j = 0; j < 4; ++j) {
                float p = __expf(s[j][r] - mnew);
                s[j][r] = p;
                rsum += p;
            }
#pragma unroll
            for (int off = 1; off < 16; off <<= 1)
                rsum += __shfl_xor(rsum, off, 32);
            lrun[r] = lrun[r] * alpha + rsum;
#pragma unroll
            for (int j = 0; j < 4; ++j) acc[j][r] *= alpha;
        }

        // ---- spill P (C layout) to wave-private LDS, re-read in A layout ----
#pragma unroll
        for (int j = 0; j < 4; ++j)
#pragma unroll
            for (int r = 0; r < 8; ++r)
                myP[(r + 8 * hi) * 64 + j * 16 + lrow] = (bf16)s[j][r];

        // ---- O += P·V ----
#pragma unroll
        for (int kc = 0; kc < 2; ++kc) {
            v16bf pa = ld_frag(myP + lrow * 64 + kc * 32 + lk);
#pragma unroll
            for (int j = 0; j < 4; ++j) {
                v16bf bv = ld_frag(sVt + (j * 16 + lrow) * 64 + kc * 32 + lk);
                acc[j] = wmma_bf16(pa, bv, acc[j]);
            }
        }
        __syncthreads();
    }

    // ---- normalize and store ctx[b, s, h*64+hd] ----
    const int b = bh >> 4, h = bh & (H_ - 1);
#pragma unroll
    for (int j = 0; j < 4; ++j)
#pragma unroll
        for (int r = 0; r < 8; ++r) {
            int rg = qb + wv * 16 + r + 8 * hi;
            float val = acc[j][r] / lrun[r];
            ctx[(((size_t)b * S_) + rg) * D_ + h * HD_ + j * 16 + lrow] = (bf16)val;
        }
}

// ---------------------------------------------------------------------------
// launcher
// ---------------------------------------------------------------------------
extern "C" void kernel_launch(void* const* d_in, const int* in_sizes, int n_in,
                              void* d_out, int out_size, void* d_ws, size_t ws_size,
                              hipStream_t stream) {
    const float* x  = (const float*)d_in[0];
    const float* Wq = (const float*)d_in[1];
    const float* Wk = (const float*)d_in[2];
    const float* Wv = (const float*)d_in[3];
    const float* Wo = (const float*)d_in[4];
    const float* bo = (const float*)d_in[5];

    const size_t nX = (size_t)B_ * S_ * D_;
    const size_t nW = (size_t)D_ * D_;

    char* ws = (char*)d_ws;
    size_t o = 0;
    bf16* xb   = (bf16*)(ws + o); o += nX * 2;
    bf16* wqb  = (bf16*)(ws + o); o += nW * 2;
    bf16* wkb  = (bf16*)(ws + o); o += nW * 2;
    bf16* wvb  = (bf16*)(ws + o); o += nW * 2;
    bf16* wob  = (bf16*)(ws + o); o += nW * 2;
    bf16* qb   = (bf16*)(ws + o); o += nX * 2;
    bf16* kb   = (bf16*)(ws + o); o += nX * 2;
    bf16* vb   = (bf16*)(ws + o); o += nX * 2;
    bf16* ctxb = (bf16*)(ws + o); o += nX * 2;   // 48 MB total

    auto cvt = [&](const float* s, bf16* d, size_t n) {
        int n4 = (int)(n / 4);
        cvt_f32_bf16<<<(n4 + 255) / 256, 256, 0, stream>>>(s, d, n4);
    };
    cvt(x,  xb,  nX);
    cvt(Wq, wqb, nW);
    cvt(Wk, wkb, nW);
    cvt(Wv, wvb, nW);
    cvt(Wo, wob, nW);

    const int M = B_ * S_;
    dim3 ggrid(M / 64, D_ / 64);
    gemm_bf16<true, false><<<ggrid, 128, 0, stream>>>(xb, wqb, nullptr, qb, M, D_, D_);
    gemm_bf16<true, false><<<ggrid, 128, 0, stream>>>(xb, wkb, nullptr, kb, M, D_, D_);
    gemm_bf16<true, false><<<ggrid, 128, 0, stream>>>(xb, wvb, nullptr, vb, M, D_, D_);

    flash_attn<<<dim3(S_ / 64, B_ * H_), 128, 0, stream>>>(qb, kb, vb, ctxb);

    gemm_bf16<false, true><<<ggrid, 128, 0, stream>>>(ctxb, wob, bo, (float*)d_out, M, D_, D_);
}
